// FullConvLayer_35794257445511
// MI455X (gfx1250) — matrix operand order repaired and checked by reading, not compiled, not run on hardware
//
#include <hip/hip_runtime.h>
#include <stdint.h>

// ---------- types ----------
typedef __bf16 bf16_t;
typedef __attribute__((ext_vector_type(16))) __bf16 v16bf;
typedef __attribute__((ext_vector_type(8)))  float  v8f;
typedef __attribute__((ext_vector_type(4)))  float  v4f;

union Frag16 { v16bf v; v4f f[2]; };
union Pack4  { bf16_t h[4]; unsigned long long u; };

__device__ __forceinline__ v4f v4zero() { v4f z; z.x = 0.f; z.y = 0.f; z.z = 0.f; z.w = 0.f; return z; }

// Problem constants
// x: [16,64,64,256]  -> upsample 2x -> xm: [16,128,128,256] (bf16, modulated)
// conv 3x3 SAME, Cout=256; out: [16,128,128,256] f32
#define BATCH 16
#define HIN   64
#define HOUT  128
#define CIN   256
#define COUT  256
#define KCHUNKS 72          // 9 taps * (256/32)
#define WPACK_PER_NT 36864  // 72 * 32 * 16 bf16 elements per cout tile

// ---------- style = dlatents @ (dk/sqrt(512)) + dense_bias ----------
__global__ void __launch_bounds__(256) style_kernel(
    const float* __restrict__ dlat, const float* __restrict__ dk,
    const float* __restrict__ dbias, float* __restrict__ style)
{
    const int b = blockIdx.x;      // 16
    const int i = threadIdx.x;     // 256
    float acc = 0.f;
    for (int j = 0; j < 512; ++j)
        acc += dlat[b * 512 + j] * dk[j * 256 + i];
    style[b * 256 + i] = acc * 0.04419417382415922f /*1/sqrt(512)*/ + dbias[i];
}

// ---------- sig[b,o] = rsqrt( sum_i style^2 * sum_tap wk^2 + 1e-8 ), wk scaled by 1/48 ----------
__global__ void __launch_bounds__(256) sig_kernel(
    const float* __restrict__ style, const float* __restrict__ ck, float* __restrict__ sig)
{
    const int b = blockIdx.x;      // 16
    const int o = threadIdx.x;     // 256
    float acc = 0.f;
    for (int i = 0; i < 256; ++i) {
        const float s = style[b * 256 + i];
        float wsum = 0.f;
        for (int t = 0; t < 9; ++t) {
            const float w = ck[((size_t)(t * 256 + i)) * 256 + o];
            wsum += w * w;
        }
        acc += s * s * wsum;
    }
    sig[b * 256 + o] = rsqrtf(acc * (1.f / 2304.f) + 1e-8f);
}

// ---------- pack weights into WMMA B-fragment layout (bf16, scaled 1/48) ----------
// B[k, n] with k = tap*256 + ci, n = cout.  Per 32-wide k-chunk:
// lane holds 16 contiguous K values (base = lane<16 ? 0 : 16) for column n = nt*16 + lane%16.
__global__ void __launch_bounds__(512) pack_kernel(
    const float* __restrict__ ck, bf16_t* __restrict__ wpack)
{
    const int t    = threadIdx.x;          // 512
    const int lane = t >> 4;               // 0..31
    const int j    = t & 15;               // 0..15
    const int blk  = blockIdx.x;           // 16*72
    const int nt   = blk / KCHUNKS;
    const int kch  = blk % KCHUNKS;
    const int tap  = kch >> 3;
    const int kc   = kch & 7;
    const int kl   = ((lane < 16) ? 0 : 16) + j;
    const int ci   = kc * 32 + kl;
    const int o    = nt * 16 + (lane & 15);
    const float w  = ck[((size_t)(tap * 256 + ci)) * 256 + o] * (1.f / 48.f); // 1/sqrt(9*256)
    wpack[(((size_t)nt * KCHUNKS + kch) * 32 + lane) * 16 + j] = (bf16_t)w;
}

// ---------- fused bilinear 2x upsample (half-pixel) + style modulation + bf16 cast ----------
__global__ void __launch_bounds__(256) upsample_mod_kernel(
    const float* __restrict__ x, const float* __restrict__ style, bf16_t* __restrict__ xm)
{
    const int b  = blockIdx.z;                       // 16
    const int y  = blockIdx.y;                       // 128
    const int c4 = threadIdx.x & 63;                 // 64 channel groups of 4
    const int xo = blockIdx.x * 4 + (threadIdx.x >> 6); // 128

    const float sy = y  * 0.5f - 0.25f;
    const float sx = xo * 0.5f - 0.25f;
    int y0 = (int)floorf(sy), x0 = (int)floorf(sx);
    const float fy = sy - (float)y0, fx = sx - (float)x0;
    const int y0c = y0 < 0 ? 0 : y0;
    const int y1c = (y0 + 1) > 63 ? 63 : (y0 + 1);
    const int x0c = x0 < 0 ? 0 : x0;
    const int x1c = (x0 + 1) > 63 ? 63 : (x0 + 1);

    const int c = c4 * 4;
    const size_t rb = (size_t)b * HIN * HIN * CIN;
    const v4f p00 = *(const v4f*)&x[rb + ((size_t)y0c * HIN + x0c) * CIN + c];
    const v4f p01 = *(const v4f*)&x[rb + ((size_t)y0c * HIN + x1c) * CIN + c];
    const v4f p10 = *(const v4f*)&x[rb + ((size_t)y1c * HIN + x0c) * CIN + c];
    const v4f p11 = *(const v4f*)&x[rb + ((size_t)y1c * HIN + x1c) * CIN + c];

    v4f v = (1.f - fy) * ((1.f - fx) * p00 + fx * p01)
          +        fy  * ((1.f - fx) * p10 + fx * p11);
    const v4f st = *(const v4f*)&style[b * 256 + c];
    v *= st;

    Pack4 o;
    o.h[0] = (bf16_t)v.x; o.h[1] = (bf16_t)v.y; o.h[2] = (bf16_t)v.z; o.h[3] = (bf16_t)v.w;
    *(unsigned long long*)&xm[(((size_t)b * HOUT + y) * HOUT + xo) * CIN + c] = o.u;
}

// ---------- main implicit-GEMM conv with WMMA bf16 ----------
// grid: (8 x-groups of 16, 4 y-groups of 32 rows, b*16+nt). block 256 = 8 waves.
// Each wave: 4 row-tiles (16 x positions each) x one 16-wide cout tile; K = 2304.
__global__ void __launch_bounds__(256) conv_wmma_kernel(
    const bf16_t* __restrict__ xm, const bf16_t* __restrict__ wpack,
    const float* __restrict__ sig, const float* __restrict__ bias,
    const float* __restrict__ noise, const float* __restrict__ nstr,
    float* __restrict__ out)
{
    __shared__ bf16_t sB[KCHUNKS * 512];   // 72 KB: full B panel for this cout tile

    const int lane = threadIdx.x & 31;
    const int wave = threadIdx.x >> 5;
    const int b    = blockIdx.z >> 4;
    const int nt   = blockIdx.z & 15;
    const int x0   = blockIdx.x * 16;
    const int ybase = blockIdx.y * 32 + wave * 4;

    // Stage the 72KB B panel into LDS with async tensor-path copies (ASYNCcnt).
    {
        const uint64_t g  = (uint64_t)(uintptr_t)(wpack + (size_t)nt * WPACK_PER_NT);
        const uint32_t l0 = (uint32_t)(uintptr_t)(&sB[0]);
        #pragma unroll
        for (int i = 0; i < 18; ++i) {
            const uint32_t off = (uint32_t)((i * 256 + threadIdx.x) * 16); // bytes
            asm volatile("global_load_async_to_lds_b128 %0, %1, off"
                         :: "v"(l0 + off), "v"(g + (uint64_t)off) : "memory");
        }
        asm volatile("s_wait_asynccnt 0" ::: "memory");
    }
    __syncthreads();

    v8f acc[4] = {};
    const int mcol  = lane & 15;            // A row (x offset) for this lane
    const int abase = (lane < 16) ? 0 : 8;  // A-frag K base / C-frag M base

    #pragma unroll 1
    for (int tap = 0; tap < 9; ++tap) {
        const int ky = tap / 3 - 1;
        const int kx = tap % 3 - 1;
        const int xx = x0 + mcol + kx;
        const bool xok = (xx >= 0) && (xx < HOUT);
        #pragma unroll
        for (int kc = 0; kc < 8; ++kc) {
            const int kch = tap * 8 + kc;
            Frag16 bf;
            const v4f* bp = reinterpret_cast<const v4f*>(&sB[kch * 512 + lane * 16]);
            bf.f[0] = bp[0];
            bf.f[1] = bp[1];
            #pragma unroll
            for (int r = 0; r < 4; ++r) {
                const int yy = ybase + r + ky;
                Frag16 af;
                if (xok && yy >= 0 && yy < HOUT) {
                    const bf16_t* ap = xm + (((size_t)(b * HOUT + yy) * HOUT + xx) << 8)
                                          + kc * 32 + (abase << 1);
                    af.f[0] = *reinterpret_cast<const v4f*>(ap);
                    af.f[1] = *reinterpret_cast<const v4f*>(ap + 8);
                } else {
                    af.f[0] = v4zero();
                    af.f[1] = v4zero();
                }
                acc[r] = __builtin_amdgcn_wmma_f32_16x16x32_bf16(
                    false, af.v, false, bf.v, (short)0, acc[r], false, false);
            }
        }
    }

    // Epilogue: demod, noise, bias, leaky-relu * sqrt(2)
    const int   co  = nt * 16 + mcol;
    const float sg  = sig[b * 256 + co];
    const float bv  = bias[co];
    const float nsv = nstr[0];
    #pragma unroll
    for (int r = 0; r < 4; ++r) {
        const int y = ybase + r;
        #pragma unroll
        for (int e = 0; e < 8; ++e) {
            const int xo = x0 + abase + e;   // C layout: M = e + (lane<16 ? 0 : 8)
            float v = acc[r][e] * sg + noise[(b * HOUT + y) * HOUT + xo] * nsv + bv;
            v = (v >= 0.f ? v : 0.2f * v) * 1.41421356237f;
            out[((((size_t)b * HOUT + y) * HOUT + xo) << 8) + co] = v;
        }
    }
}

// ---------- launch ----------
extern "C" void kernel_launch(void* const* d_in, const int* in_sizes, int n_in,
                              void* d_out, int out_size, void* d_ws, size_t ws_size,
                              hipStream_t stream) {
    (void)in_sizes; (void)n_in; (void)out_size; (void)ws_size;
    const float* x     = (const float*)d_in[0];
    const float* dlat  = (const float*)d_in[1];
    const float* ck    = (const float*)d_in[2];
    const float* dk    = (const float*)d_in[3];
    const float* dbias = (const float*)d_in[4];
    const float* bias  = (const float*)d_in[5];
    const float* noise = (const float*)d_in[6];
    const float* nstr  = (const float*)d_in[7];
    float* out = (float*)d_out;

    char* ws = (char*)d_ws;
    bf16_t* xm    = (bf16_t*)ws;                                    // 134,217,728 B
    bf16_t* wpack = (bf16_t*)(ws + (size_t)134217728);              //   1,179,648 B
    float*  style = (float*)(ws + (size_t)134217728 + 1179648);     //      16,384 B
    float*  sig   = (float*)(ws + (size_t)134217728 + 1179648 + 16384);

    style_kernel<<<dim3(BATCH), dim3(256), 0, stream>>>(dlat, dk, dbias, style);
    sig_kernel<<<dim3(BATCH), dim3(256), 0, stream>>>(style, ck, sig);
    pack_kernel<<<dim3(16 * KCHUNKS), dim3(512), 0, stream>>>(ck, wpack);
    upsample_mod_kernel<<<dim3(32, HOUT, BATCH), dim3(256), 0, stream>>>(x, style, xm);
    conv_wmma_kernel<<<dim3(8, 4, BATCH * 16), dim3(256), 0, stream>>>(
        xm, wpack, sig, bias, noise, nstr, out);
}
